// SimpleAttnBlock_11862699671779
// MI455X (gfx1250) — matrix operand order repaired and checked by reading, not compiled, not run on hardware
//
#include <hip/hip_runtime.h>

// ---------------- CDNA5 WMMA types ----------------
typedef __attribute__((ext_vector_type(16))) __bf16 v16bf;
typedef __attribute__((ext_vector_type(8)))  float  v8f;

#define Bn   4
#define Tn   1024
#define Cn   768
#define Hn   12
#define Dn   64
#define NT   (Bn * Tn)        // 4096 token rows
#define EPSF 1e-6f
#define NEG_BIG (-1e30f)

__device__ __forceinline__ unsigned short f2bf(float f) {
  union { float f; unsigned u; } v; v.f = f;
  unsigned r = v.u + 0x7FFFu + ((v.u >> 16) & 1u);   // round-nearest-even
  return (unsigned short)(r >> 16);
}

union FragU {
  uint4 q[2];
  v16bf v;
  unsigned short s[16];
};

// ---------------- 1) LayerNorm -> bf16 ----------------
__global__ __launch_bounds__(256)
void ln_bf16_kernel(const float* __restrict__ x, const float* __restrict__ g,
                    const float* __restrict__ bb, unsigned short* __restrict__ h) {
  int row = blockIdx.x;                 // 0..4095
  const float* xr = x + row * Cn;
  float s = 0.f, s2 = 0.f;
  for (int c = threadIdx.x; c < Cn; c += 256) { float v = xr[c]; s += v; s2 += v * v; }
  __shared__ float red0[8], red1[8];
  for (int m = 16; m > 0; m >>= 1) { s += __shfl_xor(s, m, 32); s2 += __shfl_xor(s2, m, 32); }
  int wave = threadIdx.x >> 5, lane = threadIdx.x & 31;
  if (lane == 0) { red0[wave] = s; red1[wave] = s2; }
  __syncthreads();
  float ts = 0.f, ts2 = 0.f;
  for (int i = 0; i < 8; i++) { ts += red0[i]; ts2 += red1[i]; }
  float mu  = ts * (1.f / Cn);
  float var = ts2 * (1.f / Cn) - mu * mu;
  float inv = rsqrtf(var + 1e-5f);
  for (int c = threadIdx.x; c < Cn; c += 256)
    h[row * Cn + c] = f2bf((xr[c] - mu) * inv * g[c] + bb[c]);
}

// ---------------- 2) transpose f32[K][N] -> bf16[N][K] ----------------
__global__ __launch_bounds__(256)
void transpose_bf16_kernel(const float* __restrict__ in, unsigned short* __restrict__ out,
                           int K, int N) {
  int idx = blockIdx.x * 256 + threadIdx.x;
  if (idx >= K * N) return;
  int k = idx / N, n = idx - k * N;
  out[n * K + k] = f2bf(in[idx]);
}

// ---------------- 3) qkv GEMM: h[4096,768] x w_attnT[2304,768] ----------------
// block tile 64(M) x 128(N); 8 waves, each 32x32 via 2x2 v_wmma_f32_16x16x32_bf16
__global__ __launch_bounds__(256)
void qkv_gemm_kernel(const unsigned short* __restrict__ A,   // h bf16 [4096][768]
                     const unsigned short* __restrict__ BT,  // w_attnT bf16 [2304][768]
                     const float* __restrict__ bias,         // [2304]
                     unsigned short* __restrict__ q,
                     unsigned short* __restrict__ k,
                     unsigned short* __restrict__ v) {
  __shared__ __align__(16) unsigned short As[64][32];
  __shared__ __align__(16) unsigned short Bs[128][32];
  int mb = blockIdx.x * 64, nb = blockIdx.y * 128;
  int tid = threadIdx.x, lane = tid & 31, wave = tid >> 5;
  int wm = wave & 1, wn = wave >> 1;   // wm in [0,2), wn in [0,4)
  v8f acc[2][2] = {};
  int arow0 = wm * 32 + (lane & 15);
  int akb   = (lane < 16) ? 0 : 8;     // ISA 16-bit A layout
  int brow0 = wn * 32 + (lane & 15);
  int bkb   = (lane < 16) ? 0 : 8;     // ISA 16-bit B layout

  for (int k0 = 0; k0 < Cn; k0 += 32) {
    __syncthreads();
    { int r = tid >> 2, cg = (tid & 3) * 8;
      *(uint4*)&As[r][cg] = *(const uint4*)&A[(mb + r) * Cn + k0 + cg];
      if (k0 + 32 < Cn) __builtin_prefetch(&A[(mb + r) * Cn + k0 + 32 + cg], 0, 0); }
    { int r = tid >> 1, cg = (tid & 1) * 16;
      *(uint4*)&Bs[r][cg]     = *(const uint4*)&BT[(nb + r) * Cn + k0 + cg];
      *(uint4*)&Bs[r][cg + 8] = *(const uint4*)&BT[(nb + r) * Cn + k0 + cg + 8];
      if (k0 + 32 < Cn) __builtin_prefetch(&BT[(nb + r) * Cn + k0 + 32 + cg], 0, 0); }
    __syncthreads();

    FragU af[2], bf2[2];
    af[0].q[0] = *(const uint4*)&As[arow0][akb];
    af[0].q[1] = *(const uint4*)&As[arow0][akb + 16];
    af[1].q[0] = *(const uint4*)&As[arow0 + 16][akb];
    af[1].q[1] = *(const uint4*)&As[arow0 + 16][akb + 16];
    bf2[0].q[0] = *(const uint4*)&Bs[brow0][bkb];
    bf2[0].q[1] = *(const uint4*)&Bs[brow0][bkb + 16];
    bf2[1].q[0] = *(const uint4*)&Bs[brow0 + 16][bkb];
    bf2[1].q[1] = *(const uint4*)&Bs[brow0 + 16][bkb + 16];
#pragma unroll
    for (int i = 0; i < 2; i++)
#pragma unroll
      for (int j = 0; j < 2; j++)
        acc[i][j] = __builtin_amdgcn_wmma_f32_16x16x32_bf16(
            false, af[i].v, false, bf2[j].v, (short)0, acc[i][j], false, false);
  }

#pragma unroll
  for (int i = 0; i < 2; i++)
#pragma unroll
    for (int j = 0; j < 2; j++) {
      int n = nb + wn * 32 + j * 16 + (lane & 15);
      float bv = bias[n];
      int which = n / Cn, c = n - which * Cn;
      float scale = (which == 0) ? 0.125f : 1.0f;       // sm_scale folded into q
      unsigned short* dst = (which == 0) ? q : (which == 1) ? k : v;
      int head = c >> 6, d = c & 63;
#pragma unroll
      for (int r = 0; r < 8; r++) {
        int m = mb + wm * 32 + i * 16 + r + ((lane >> 4) << 3);
        int bi = m >> 10, t = m & 1023;
        dst[((((bi * Hn + head) << 10) + t) << 6) + d] = f2bf((acc[i][j][r] + bv) * scale);
      }
    }
}

// ---------------- 4) Stieltjes attention: full causal score panel in 320KB LDS ----------------
__global__ __launch_bounds__(256)
void stieltjes_attn_kernel(const unsigned short* __restrict__ Q,
                           const unsigned short* __restrict__ K,
                           const unsigned short* __restrict__ V,
                           unsigned short* __restrict__ O) {   // [4096][768]
  __shared__ __align__(16) float S[64][Tn];                    // 256 KB score panel
  __shared__ __align__(16) unsigned short qt[64][64];          // 8 KB
  __shared__ __align__(16) unsigned short kt[64][64];          // 8 KB (reused as vT)
  __shared__ float zrow[64], isrow[64];

  int qb = blockIdx.x, h = blockIdx.y, bi = blockIdx.z;
  int tid = threadIdx.x, lane = tid & 31, wave = tid >> 5;
  int mi = wave & 3, nj = wave >> 2;     // mi: 16-row slab, nj: 32-col half
  size_t headoff = (size_t)((bi * Hn + h) << 10) << 6;   // (b*H+h)*T*D
  const unsigned short* Qh = Q + headoff;
  const unsigned short* Kh = K + headoff;
  const unsigned short* Vh = V + headoff;

  // load q tile [64][64]
  { int r = tid >> 2, cg = (tid & 3) * 16;
    *(uint4*)&qt[r][cg]     = *(const uint4*)&Qh[((qb << 6) + r) * Dn + cg];
    *(uint4*)&qt[r][cg + 8] = *(const uint4*)&Qh[((qb << 6) + r) * Dn + cg + 8]; }
  __syncthreads();

  // q A-fragments (invariant over key blocks): rows mi*16, K chunks d=0 and d=32
  FragU aq[2];
  { int arow = mi * 16 + (lane & 15);
    int akb = (lane < 16) ? 0 : 8;
    aq[0].q[0] = *(const uint4*)&qt[arow][akb];
    aq[0].q[1] = *(const uint4*)&qt[arow][akb + 16];
    aq[1].q[0] = *(const uint4*)&qt[arow][32 + akb];
    aq[1].q[1] = *(const uint4*)&qt[arow][32 + akb + 16]; }

  // ---- phase 1: causal score panel via WMMA ----
  for (int kb = 0; kb <= qb; kb++) {
    __syncthreads();
    { int r = tid >> 2, cg = (tid & 3) * 16;
      *(uint4*)&kt[r][cg]     = *(const uint4*)&Kh[((kb << 6) + r) * Dn + cg];
      *(uint4*)&kt[r][cg + 8] = *(const uint4*)&Kh[((kb << 6) + r) * Dn + cg + 8]; }
    __syncthreads();
#pragma unroll
    for (int tt = 0; tt < 2; tt++) {
      int jrow = nj * 32 + tt * 16 + (lane & 15);
      int bkb = (lane < 16) ? 0 : 8;
      FragU b0, b1;
      b0.q[0] = *(const uint4*)&kt[jrow][bkb];
      b0.q[1] = *(const uint4*)&kt[jrow][bkb + 16];
      b1.q[0] = *(const uint4*)&kt[jrow][32 + bkb];
      b1.q[1] = *(const uint4*)&kt[jrow][32 + bkb + 16];
      v8f sc = {};
      sc = __builtin_amdgcn_wmma_f32_16x16x32_bf16(false, aq[0].v, false, b0.v, (short)0, sc, false, false);
      sc = __builtin_amdgcn_wmma_f32_16x16x32_bf16(false, aq[1].v, false, b1.v, (short)0, sc, false, false);
      int coll = nj * 32 + tt * 16 + (lane & 15);
      int jg = (kb << 6) + coll;
#pragma unroll
      for (int r = 0; r < 8; r++) {
        int rowl = mi * 16 + r + ((lane >> 4) << 3);
        int ig = (qb << 6) + rowl;
        S[rowl][(kb << 6) + coll] = (jg <= ig) ? sc[r] : NEG_BIG;
      }
    }
  }
  __syncthreads();

  // ---- phase 2: per-row Newton iterations over the in-LDS score row ----
  int jmax = (qb + 1) << 6;
  for (int rr = 0; rr < 8; rr++) {
    int rowl = wave * 8 + rr;
    const float* srow = S[rowl];
    float smax = NEG_BIG;
    for (int j = lane; j < jmax; j += 32) smax = fmaxf(smax, srow[j]);
    for (int m = 16; m; m >>= 1) smax = fmaxf(smax, __shfl_xor(smax, m, 32));
    float z = smax + 1.0f;
#pragma unroll
    for (int it = 0; it < 5; it++) {
      float s1 = 0.f, s2 = 0.f;
      for (int j = lane; j < jmax; j += 32) {
        float d = fmaxf(z - srow[j], EPSF);
        float r = 1.0f / d;                 // SQ==1: w = d^-1, s2 term = d^-2
        s1 += r; s2 += r * r;
      }
      for (int m = 16; m; m >>= 1) { s1 += __shfl_xor(s1, m, 32); s2 += __shfl_xor(s2, m, 32); }
      z = fmaxf(z + (s1 - 1.0f) / (s2 + EPSF), smax + EPSF);
    }
    float sw = 0.f;
    for (int j = lane; j < jmax; j += 32) sw += 1.0f / fmaxf(z - srow[j], EPSF);
    for (int m = 16; m; m >>= 1) sw += __shfl_xor(sw, m, 32);
    if (lane == 0) { zrow[rowl] = z; isrow[rowl] = 1.0f / (sw + EPSF); }
  }
  __syncthreads();

  // ---- phase 3: o = p @ v via WMMA; p regenerated from (z, invsum, S) ----
  v8f oacc[2] = {};
  for (int kb = 0; kb <= qb; kb++) {
    __syncthreads();
    { // load v block transposed into kt as vT[d][j]
      int jr = tid >> 2, cg = (tid & 3) * 16;
      uint4 x0 = *(const uint4*)&Vh[((kb << 6) + jr) * Dn + cg];
      uint4 x1 = *(const uint4*)&Vh[((kb << 6) + jr) * Dn + cg + 8];
      const unsigned short* px = (const unsigned short*)&x0;
#pragma unroll
      for (int e = 0; e < 8; e++) kt[cg + e][jr] = px[e];
      px = (const unsigned short*)&x1;
#pragma unroll
      for (int e = 0; e < 8; e++) kt[cg + 8 + e][jr] = px[e];
    }
    __syncthreads();
#pragma unroll
    for (int kc = 0; kc < 2; kc++) {
      int prow = mi * 16 + (lane & 15);
      float z = zrow[prow], isv = isrow[prow];
      int jb = (kb << 6) + kc * 32 + ((lane < 16) ? 0 : 8);
      FragU pa;
#pragma unroll
      for (int e = 0; e < 8; e++)
        pa.s[e] = f2bf(isv / fmaxf(z - S[prow][jb + e], EPSF));
#pragma unroll
      for (int e = 0; e < 8; e++)
        pa.s[8 + e] = f2bf(isv / fmaxf(z - S[prow][jb + 16 + e], EPSF));
#pragma unroll
      for (int tt = 0; tt < 2; tt++) {
        int drow = nj * 32 + tt * 16 + (lane & 15);
        int jcb = kc * 32 + ((lane < 16) ? 0 : 8);
        FragU vb;
        vb.q[0] = *(const uint4*)&kt[drow][jcb];
        vb.q[1] = *(const uint4*)&kt[drow][jcb + 16];
        oacc[tt] = __builtin_amdgcn_wmma_f32_16x16x32_bf16(
            false, pa.v, false, vb.v, (short)0, oacc[tt], false, false);
      }
    }
  }

#pragma unroll
  for (int tt = 0; tt < 2; tt++) {
    int d = nj * 32 + tt * 16 + (lane & 15);
    int col = (h << 6) + d;
#pragma unroll
    for (int r = 0; r < 8; r++) {
      int rowl = mi * 16 + r + ((lane >> 4) << 3);
      int tok = (bi << 10) + (qb << 6) + rowl;
      O[tok * Cn + col] = f2bf(oacc[tt][r]);
    }
  }
}

// ---------------- 5) proj GEMM + bias + residual ----------------
__global__ __launch_bounds__(256)
void proj_gemm_kernel(const unsigned short* __restrict__ A,   // o bf16 [4096][768]
                      const unsigned short* __restrict__ BT,  // w_projT bf16 [768][768]
                      const float* __restrict__ bias,
                      const float* __restrict__ x,
                      float* __restrict__ out) {
  __shared__ __align__(16) unsigned short As[64][32];
  __shared__ __align__(16) unsigned short Bs[128][32];
  int mb = blockIdx.x * 64, nb = blockIdx.y * 128;
  int tid = threadIdx.x, lane = tid & 31, wave = tid >> 5;
  int wm = wave & 1, wn = wave >> 1;
  v8f acc[2][2] = {};
  int arow0 = wm * 32 + (lane & 15);
  int akb   = (lane < 16) ? 0 : 8;
  int brow0 = wn * 32 + (lane & 15);
  int bkb   = (lane < 16) ? 0 : 8;

  for (int k0 = 0; k0 < Cn; k0 += 32) {
    __syncthreads();
    { int r = tid >> 2, cg = (tid & 3) * 8;
      *(uint4*)&As[r][cg] = *(const uint4*)&A[(mb + r) * Cn + k0 + cg];
      if (k0 + 32 < Cn) __builtin_prefetch(&A[(mb + r) * Cn + k0 + 32 + cg], 0, 0); }
    { int r = tid >> 1, cg = (tid & 1) * 16;
      *(uint4*)&Bs[r][cg]     = *(const uint4*)&BT[(nb + r) * Cn + k0 + cg];
      *(uint4*)&Bs[r][cg + 8] = *(const uint4*)&BT[(nb + r) * Cn + k0 + cg + 8]; }
    __syncthreads();

    FragU af[2], bf2[2];
    af[0].q[0] = *(const uint4*)&As[arow0][akb];
    af[0].q[1] = *(const uint4*)&As[arow0][akb + 16];
    af[1].q[0] = *(const uint4*)&As[arow0 + 16][akb];
    af[1].q[1] = *(const uint4*)&As[arow0 + 16][akb + 16];
    bf2[0].q[0] = *(const uint4*)&Bs[brow0][bkb];
    bf2[0].q[1] = *(const uint4*)&Bs[brow0][bkb + 16];
    bf2[1].q[0] = *(const uint4*)&Bs[brow0 + 16][bkb];
    bf2[1].q[1] = *(const uint4*)&Bs[brow0 + 16][bkb + 16];
#pragma unroll
    for (int i = 0; i < 2; i++)
#pragma unroll
      for (int j = 0; j < 2; j++)
        acc[i][j] = __builtin_amdgcn_wmma_f32_16x16x32_bf16(
            false, af[i].v, false, bf2[j].v, (short)0, acc[i][j], false, false);
  }

#pragma unroll
  for (int i = 0; i < 2; i++)
#pragma unroll
    for (int j = 0; j < 2; j++) {
      int n = nb + wn * 32 + j * 16 + (lane & 15);
      float bv = bias[n];
#pragma unroll
      for (int r = 0; r < 8; r++) {
        int m = mb + wm * 32 + i * 16 + r + ((lane >> 4) << 3);
        out[m * Cn + n] = acc[i][j][r] + bv + x[m * Cn + n];
      }
    }
}

// ---------------- host launcher ----------------
extern "C" void kernel_launch(void* const* d_in, const int* in_sizes, int n_in,
                              void* d_out, int out_size, void* d_ws, size_t ws_size,
                              hipStream_t stream) {
  (void)in_sizes; (void)n_in; (void)out_size; (void)ws_size;
  const float* x      = (const float*)d_in[0];
  const float* w_attn = (const float*)d_in[1];
  const float* b_attn = (const float*)d_in[2];
  const float* w_proj = (const float*)d_in[3];
  const float* b_proj = (const float*)d_in[4];
  const float* ln_g   = (const float*)d_in[5];
  const float* ln_b   = (const float*)d_in[6];
  float* out = (float*)d_out;

  char* ws = (char*)d_ws;
  size_t off = 0;
  auto wsalloc = [&](size_t bytes) -> void* {
    void* p = (void*)(ws + off);
    off += (bytes + 255) & ~(size_t)255;
    return p;
  };
  unsigned short* h_bf   = (unsigned short*)wsalloc((size_t)NT * Cn * 2);
  unsigned short* wattnT = (unsigned short*)wsalloc((size_t)3 * Cn * Cn * 2);
  unsigned short* wprojT = (unsigned short*)wsalloc((size_t)Cn * Cn * 2);
  unsigned short* q_bf   = (unsigned short*)wsalloc((size_t)NT * Cn * 2);
  unsigned short* k_bf   = (unsigned short*)wsalloc((size_t)NT * Cn * 2);
  unsigned short* v_bf   = (unsigned short*)wsalloc((size_t)NT * Cn * 2);
  unsigned short* o_bf   = (unsigned short*)wsalloc((size_t)NT * Cn * 2);

  ln_bf16_kernel<<<NT, 256, 0, stream>>>(x, ln_g, ln_b, h_bf);
  transpose_bf16_kernel<<<(Cn * 3 * Cn + 255) / 256, 256, 0, stream>>>(w_attn, wattnT, Cn, 3 * Cn);
  transpose_bf16_kernel<<<(Cn * Cn + 255) / 256, 256, 0, stream>>>(w_proj, wprojT, Cn, Cn);
  qkv_gemm_kernel<<<dim3(NT / 64, (3 * Cn) / 128), 256, 0, stream>>>(
      h_bf, wattnT, b_attn, q_bf, k_bf, v_bf);
  stieltjes_attn_kernel<<<dim3(Tn / 64, Hn, Bn), 256, 0, stream>>>(q_bf, k_bf, v_bf, o_bf);
  proj_gemm_kernel<<<dim3(NT / 64, Cn / 128), 256, 0, stream>>>(o_bf, wprojT, b_proj, x, out);
}
